// ScaledDotProductAttention_22625887715975
// MI455X (gfx1250) — matrix operand order repaired and checked by reading, not compiled, not run on hardware
//
#include <hip/hip_runtime.h>

typedef __attribute__((ext_vector_type(2)))  float    v2f;
typedef __attribute__((ext_vector_type(8)))  float    v8f;
typedef __attribute__((ext_vector_type(16))) _Float16 v16h;

#define BH   64
#define NSEQ 1024
#define DK   64
#define RPB  32          // rows of Q handled per workgroup
#define SSTR 1028        // score-tile LDS stride (floats), 1028%64=4 -> bank-spread
#define QSTR 68          // qf tile stride (floats)
#define KSTR 68          // kf chunk stride (floats)
#define VSTR 72          // vT chunk stride (halves)
#define NTHREADS 256

// dynamic LDS layout (bytes):
//  sS   : 32*1028*4 = 131584   score/prob stripe (resident across phases)
//  sQ   : 32*68*4   =   8704   fused Q tile
//  sK   : 64*68*4   =  17408   fused K chunk (phase 1) / union with vT f16 (phase 2)
//  sRed : 32*8*4    =   1024   partial reductions
//  sMax/sSum: 64*4  =    256
#define SMEM_BYTES (131584 + 8704 + 17408 + 1024 + 256)

__global__ __launch_bounds__(NTHREADS)
void attn_sem_kernel(const float* __restrict__ q,  const float* __restrict__ k,
                     const float* __restrict__ v,  const float* __restrict__ qs,
                     const float* __restrict__ ks,
                     float* __restrict__ out, float* __restrict__ attn,
                     float* __restrict__ score)
{
    extern __shared__ char smem_raw[];
    float*    sS   = (float*)smem_raw;            // [RPB][SSTR]
    float*    sQ   = sS + RPB * SSTR;             // [RPB][QSTR]
    float*    sK   = sQ + RPB * QSTR;             // [64][KSTR]
    _Float16* sV   = (_Float16*)sK;               // [64 dims][VSTR] (union with sK)
    float*    sRed = sK + 64 * KSTR;              // [RPB][8]
    float*    sMax = sRed + RPB * 8;              // [RPB]
    float*    sSum = sMax + RPB;                  // [RPB]

    const int t    = threadIdx.x;
    const int wave = t >> 5;
    const int lane = t & 31;
    const int l15  = lane & 15;
    const int hi   = lane >> 4;

    const int  batch  = blockIdx.x >> 5;          // 32 row-stripes per batch
    const int  row0   = (blockIdx.x & 31) * RPB;
    const long inBase = (long)batch * NSEQ * DK;

    // ---------------- load fused Q stripe: qf = q + q_sem ----------------
    {
        const float* qB  = q  + inBase + (long)row0 * DK;
        const float* qsB = qs + inBase + (long)row0 * DK;
        #pragma unroll
        for (int i = 0; i < (RPB * DK) / NTHREADS; ++i) {
            int e = t + NTHREADS * i;
            sQ[(e >> 6) * QSTR + (e & 63)] = qB[e] + qsB[e];
        }
    }
    __syncthreads();

    // ---------------- phase 1: S = qf @ kf^T / 8 (fp32 WMMA) ----------------
    const int rt  = wave & 1;   // 16-row tile within stripe
    const int ctl = wave >> 1;  // 16-col tile within 64-col chunk (0..3)
    for (int ch = 0; ch < 16; ++ch) {
        const int c0 = ch * 64;
        // cooperative load of fused K chunk: kf = k + k_sem
        #pragma unroll
        for (int i = 0; i < (64 * DK) / NTHREADS; ++i) {
            int e = t + NTHREADS * i;
            int r = e >> 6, d = e & 63;
            long g = inBase + (long)(c0 + r) * DK + d;
            sK[r * KSTR + d] = k[g] + ks[g];
        }
        __syncthreads();

        v8f acc = {};
        // A: lane in 0..15 -> M, VGPR j -> K = j + 2*hi   (ISA 7.12.2, 32-bit A 16x4)
        // B: lane in 0..15 -> N, VGPR j -> K = j + 2*hi   (mirror layout)
        const float* aRow = sQ + (l15 + 16 * rt) * QSTR + 2 * hi;
        const float* bRow = sK + (16 * ctl + l15) * KSTR + 2 * hi;
        #pragma unroll
        for (int kk = 0; kk < DK; kk += 4) {
            v2f a = *(const v2f*)(aRow + kk);
            v2f b = *(const v2f*)(bRow + kk);
            acc = __builtin_amdgcn_wmma_f32_16x16x4_f32(
                      false, a, false, b, (short)0, acc, false, false);
        }
        // D layout: VGPR r, lane -> (M = r + 8*hi, N = l15); scale by 1/TEMPERATURE
        #pragma unroll
        for (int r = 0; r < 8; ++r) {
            int m = 16 * rt + r + 8 * hi;
            sS[m * SSTR + c0 + 16 * ctl + l15] = acc[r] * 0.125f;
        }
        __syncthreads();
    }

    // ---------------- softmax over resident stripe ----------------
    {   // partial row max: 8 threads per row
        int r = t >> 3, g = t & 7;
        const float* row = sS + r * SSTR;
        float mx = -3.4e38f;
        for (int c = g; c < NSEQ; c += 8) mx = fmaxf(mx, row[c]);
        sRed[r * 8 + g] = mx;
    }
    __syncthreads();
    if (t < RPB) {
        float mx = sRed[t * 8];
        #pragma unroll
        for (int g = 1; g < 8; ++g) mx = fmaxf(mx, sRed[t * 8 + g]);
        sMax[t] = mx;
    }
    __syncthreads();
    {   // partial row sum of exp
        int r = t >> 3, g = t & 7;
        const float* row = sS + r * SSTR;
        float mx = sMax[r], sum = 0.f;
        for (int c = g; c < NSEQ; c += 8) sum += __expf(row[c] - mx);
        sRed[r * 8 + g] = sum;
    }
    __syncthreads();
    if (t < RPB) {
        float s2 = 0.f;
        #pragma unroll
        for (int g = 0; g < 8; ++g) s2 += sRed[t * 8 + g];
        sSum[t] = 1.0f / s2;
    }
    __syncthreads();

    // ---- write attn_score + attn (coalesced, one wave per row), p -> sS ----
    {
        float* scoreB = score + (long)batch * NSEQ * NSEQ + (long)row0 * NSEQ;
        float* attnB  = attn  + (long)batch * NSEQ * NSEQ + (long)row0 * NSEQ;
        #pragma unroll
        for (int pass = 0; pass < 4; ++pass) {
            int r  = wave + pass * 8;
            float mx = sMax[r], inv = sSum[r];
            float* row = sS + r * SSTR;
            long gb = (long)r * NSEQ;
            for (int c = lane; c < NSEQ; c += 32) {
                float sc = row[c];
                float p  = __expf(sc - mx) * inv;
                scoreB[gb + c] = sc;
                attnB [gb + c] = p;
                row[c] = p;                 // stripe now holds probabilities
            }
        }
    }
    __syncthreads();

    // ---------------- phase 2: O = P @ V (f16 WMMA, fp32 accum) ----------------
    const int nt = wave >> 1;   // 16-wide dim tile (0..3); rt reused for row tile
    v8f oacc = {};
    for (int ch = 0; ch < 16; ++ch) {
        const int k0 = ch * 64;
        __syncthreads();        // previous chunk's vT fully consumed
        // load V chunk transposed into LDS as f16: sV[dim][kk]
        #pragma unroll
        for (int i = 0; i < (64 * DK) / NTHREADS; ++i) {
            int e = t + NTHREADS * i;
            int kr = e >> 6, d = e & 63;
            sV[d * VSTR + kr] = (_Float16)v[inBase + (long)(k0 + kr) * DK + d];
        }
        __syncthreads();

        #pragma unroll
        for (int sub = 0; sub < 2; ++sub) {
            // A (16x32 f16): VGPR j<4 -> K = 2j+8*hi ; j>=4 -> K = 16+2(j-4)+8*hi
            v16h A;
            const float* pRow = sS + (l15 + 16 * rt) * SSTR + k0 + sub * 32;
            #pragma unroll
            for (int j = 0; j < 8; ++j) {
                int kOff = (j < 4) ? (2 * j + 8 * hi) : (16 + 2 * (j - 4) + 8 * hi);
                v2f pv = *(const v2f*)(pRow + kOff);
                A[2 * j]     = (_Float16)pv[0];
                A[2 * j + 1] = (_Float16)pv[1];
            }
            // B (32x16 f16): lane -> N, VGPR j<4 -> K = 2j+8*hi ; j>=4 -> K = 16+2(j-4)+8*hi
            v16h Bv;
            const _Float16* vRow = sV + (16 * nt + l15) * VSTR + sub * 32;
            #pragma unroll
            for (int j = 0; j < 8; ++j) {
                int kOff = (j < 4) ? (2 * j + 8 * hi) : (16 + 2 * (j - 4) + 8 * hi);
                Bv[2 * j]     = vRow[kOff];
                Bv[2 * j + 1] = vRow[kOff + 1];
            }
            oacc = __builtin_amdgcn_wmma_f32_16x16x32_f16(
                       false, A, false, Bv, (short)0, oacc, false, false);
        }
    }
    // write O tile: (M = 16*rt + r + 8*hi, dim = 16*nt + l15)
    {
        float* outB = out + inBase + (long)row0 * DK;
        #pragma unroll
        for (int r = 0; r < 8; ++r) {
            int m = 16 * rt + r + 8 * hi;
            outB[(long)m * DK + 16 * nt + l15] = oacc[r];
        }
    }
}

extern "C" void kernel_launch(void* const* d_in, const int* in_sizes, int n_in,
                              void* d_out, int out_size, void* d_ws, size_t ws_size,
                              hipStream_t stream) {
    const float* q  = (const float*)d_in[0];
    const float* k  = (const float*)d_in[1];
    const float* v  = (const float*)d_in[2];
    const float* qs = (const float*)d_in[3];
    const float* ks = (const float*)d_in[4];

    float* out   = (float*)d_out;                               // [64,1024,64]
    float* attn  = out  + (long)BH * NSEQ * DK;                 // [64,1024,1024]
    float* score = attn + (long)BH * NSEQ * NSEQ;               // [64,1024,1024]

    (void)hipFuncSetAttribute((const void*)attn_sem_kernel,
                              hipFuncAttributeMaxDynamicSharedMemorySize,
                              SMEM_BYTES);

    dim3 grid(BH * (NSEQ / RPB));   // 2048 workgroups
    dim3 block(NTHREADS);           // 8 wave32
    attn_sem_kernel<<<grid, block, SMEM_BYTES, stream>>>(q, k, v, qs, ks,
                                                         out, attn, score);
}
